// DecoderAttention_21921513079147
// MI455X (gfx1250) — matrix operand order repaired and checked by reading, not compiled, not run on hardware
//
#include <hip/hip_runtime.h>
#include <hip/hip_bf16.h>

typedef __bf16 bf16;
typedef __attribute__((ext_vector_type(16))) __bf16 v16bf;
typedef __attribute__((ext_vector_type(8)))  __bf16 v8bf;
typedef __attribute__((ext_vector_type(4)))  __bf16 v4bf;
typedef __attribute__((ext_vector_type(8)))  float   v8f;

#define NB   4
#define NT   2048
#define CCH  1024
#define NH   16
#define DH   64
#define QSCALE 0.08838834764831845f   // 1/sqrt(2*D) = 1/sqrt(128)

static __device__ __forceinline__ v8f wmma_bf16(v16bf a, v16bf b, v8f c) {
  // D = A(16x32 bf16) * B(32x16 bf16) + C(16x16 f32)
  return __builtin_amdgcn_wmma_f32_16x16x32_bf16(false, a, false, b, (short)0, c,
                                                 false, false);
}

// A fragment: 16x32 tile, row-major, leading dim `ld` (elements).
// lane l: row m = l&15, k-base = (l<16)?0:8, elems k = kb+{0..7} U kb+{16..23}
static __device__ __forceinline__ v16bf frag_a_ld(const bf16* base, int ld) {
  const int l  = threadIdx.x & 31;
  const int m  = l & 15;
  const int kb = (l < 16) ? 0 : 8;
  const bf16* p = base + m * ld + kb;
  v8bf lo = *(const v8bf*)(p);
  v8bf hi = *(const v8bf*)(p + 16);
  return __builtin_shufflevector(lo, hi, 0,1,2,3,4,5,6,7,8,9,10,11,12,13,14,15);
}

// B fragment: 32x16 tile stored "n-major": row n holds the 32 k-values.
// lane l: col n = l&15, k-base = (l<16)?0:8, elems k = kb+{0..7} U kb+{16..23}
static __device__ __forceinline__ v16bf frag_b_ld(const bf16* base, int ld) {
  const int l  = threadIdx.x & 31;
  const int n  = l & 15;
  const int kb = (l < 16) ? 0 : 8;
  const bf16* p = base + n * ld + kb;
  v8bf lo = *(const v8bf*)(p);
  v8bf hi = *(const v8bf*)(p + 16);
  return __builtin_shufflevector(lo, hi, 0,1,2,3,4,5,6,7,8,9,10,11,12,13,14,15);
}

// CDNA5 async global->LDS copy, 16B per lane, tracked by ASYNCcnt.
static __device__ __forceinline__ void async_copy_b128(const bf16* g, bf16* lds) {
  const unsigned laddr = (unsigned)(unsigned long long)lds;  // low 32b = LDS offset
  asm volatile("global_load_async_to_lds_b128 %0, %1, off"
               :: "v"(laddr), "v"(g) : "memory");
}
static __device__ __forceinline__ void async_wait0() {
  asm volatile("s_wait_asynccnt 0" ::: "memory");
}
static __device__ __forceinline__ void async_wait6() {
  asm volatile("s_wait_asynccnt 6" ::: "memory");
}

// ---------------------------------------------------------------------------
// Kernel 0: f32 -> bf16 conversion (one-time; keeps GEMM hot loop cvt-free)
// ---------------------------------------------------------------------------
__global__ __launch_bounds__(256) void cvt_bf16_kernel(
    const float* __restrict__ src, bf16* __restrict__ dst, int n4) {
  const int i = blockIdx.x * 256 + threadIdx.x;
  if (i >= n4) return;
  float4 v = ((const float4*)src)[i];
  v4bf o = {(bf16)v.x, (bf16)v.y, (bf16)v.z, (bf16)v.w};
  ((v4bf*)dst)[i] = o;
}

// ---------------------------------------------------------------------------
// Kernel 1: QKV projection  Y[t,n] = sum_k X[t,k]*W[n,k]   (y = x @ W^T)
// 64(M) x 128(N) tile per block, 256 threads = 8 waves (4 row-groups x 2 col
// halves), K-step 64, double-buffered async global->LDS tile fills.
// All 10 fragment loads of a K-step are hoisted ahead of the 8 WMMAs so the
// LDS clause drains once instead of stalling each WMMA on s_wait_dscnt.
// z=0 -> Q [B,H,T,D] (scaled), z=1 -> K [B,H,T,D], z=2 -> V^T [B,H,D,T]
// ---------------------------------------------------------------------------
__global__ __launch_bounds__(256) void qkv_gemm_kernel(
    const bf16* __restrict__ xb, const bf16* __restrict__ wb,
    bf16* __restrict__ q, bf16* __restrict__ k, bf16* __restrict__ vt) {
  __shared__ bf16 lA[2][64 * 64];    //  8 KB per buffer
  __shared__ bf16 lB[2][128 * 64];   // 16 KB per buffer

  const int z = blockIdx.z;
  const bf16* __restrict__ W = wb + (size_t)z * CCH * CCH;
  const int Nbase = blockIdx.x * 128;
  const int Mbase = blockIdx.y * 64;
  const int tid    = threadIdx.x;
  const int w      = tid >> 5;
  const int lane   = tid & 31;
  const int n16    = lane & 15;
  const int half   = lane >> 4;
  const int rowgrp = w & 3;   // 16-row group within the 64-row tile
  const int colgrp = w >> 2;  // 64-col half within the 128-col tile

  const v8f vzero = {0.f,0.f,0.f,0.f,0.f,0.f,0.f,0.f};
  v8f acc[4];
#pragma unroll
  for (int f = 0; f < 4; ++f) acc[f] = vzero;

  // issue one K-step tile (A: 64x64, B: 128x64) as async 16B copies;
  // 6 async ops per thread per tile.
  auto issue_tile = [&](int kb, int buf) {
#pragma unroll
    for (int i = 0; i < 2; ++i) {
      const int chunk = tid + 256 * i;           // 0..511
      const int row = chunk >> 3;                // 0..63
      const int c8  = (chunk & 7) * 8;           // 0..56
      async_copy_b128(xb + (size_t)(Mbase + row) * CCH + kb + c8,
                      &lA[buf][row * 64 + c8]);
    }
#pragma unroll
    for (int i = 0; i < 4; ++i) {
      const int chunk = tid + 256 * i;           // 0..1023
      const int row = chunk >> 3;                // 0..127
      const int c8  = (chunk & 7) * 8;
      async_copy_b128(W + (size_t)(Nbase + row) * CCH + kb + c8,
                      &lB[buf][row * 64 + c8]);
    }
  };

  issue_tile(0, 0);
  const int KSTEPS = CCH / 64;   // 16
  for (int ks = 0; ks < KSTEPS; ++ks) {
    const int buf = ks & 1;
    if (ks + 1 < KSTEPS) {
      issue_tile((ks + 1) * 64, buf ^ 1);
      async_wait6();             // drain only the current tile's 6 ops
    } else {
      async_wait0();
    }
    __syncthreads();
    // hoist all fragment loads, then run the 8 WMMAs back-to-back
    v16bf a0 = frag_a_ld(&lA[buf][(16 * rowgrp) * 64], 64);
    v16bf a1 = frag_a_ld(&lA[buf][(16 * rowgrp) * 64 + 32], 64);
    v16bf bfr[8];
#pragma unroll
    for (int f = 0; f < 4; ++f) {
      const bf16* bbase = &lB[buf][(64 * colgrp + 16 * f) * 64];
      bfr[2 * f]     = frag_b_ld(bbase, 64);
      bfr[2 * f + 1] = frag_b_ld(bbase + 32, 64);
    }
#pragma unroll
    for (int f = 0; f < 4; ++f) {
      acc[f] = wmma_bf16(a0, bfr[2 * f],     acc[f]);
      acc[f] = wmma_bf16(a1, bfr[2 * f + 1], acc[f]);
    }
    __syncthreads();
  }

#pragma unroll
  for (int f = 0; f < 4; ++f) {
#pragma unroll
    for (int r = 0; r < 8; ++r) {
      float val = acc[f][r];
      const int tlin = Mbase + 16 * rowgrp + r + 8 * half;
      const int b    = tlin >> 11;          // / NT
      const int tt   = tlin & (NT - 1);
      const int c    = Nbase + 64 * colgrp + 16 * f + n16;
      const int h    = c >> 6;
      const int d    = c & 63;
      const size_t bh = (size_t)(b * NH + h);
      if (z == 0) {
        q[(bh * NT + tt) * DH + d] = (bf16)(val * QSCALE);
      } else if (z == 1) {
        k[(bh * NT + tt) * DH + d] = (bf16)val;
      } else {
        vt[(bh * DH + d) * NT + tt] = (bf16)val;
      }
    }
  }
}

// ---------------------------------------------------------------------------
// Kernel 2: causal flash attention. One wave = 16 query rows, key blocks of 64.
// S = Q K^T (already scaled via Q), online softmax, O += P V.
// K/V^T B-fragments are two 16B per-lane global loads each (no transpose).
// V fragment loads are issued right after the S-WMMAs so global latency is
// hidden under the softmax VALU work.
// ---------------------------------------------------------------------------
__global__ __launch_bounds__(128) void flash_attn_kernel(
    const bf16* __restrict__ Q, const bf16* __restrict__ K,
    const bf16* __restrict__ Vt, float* __restrict__ attn) {
  __shared__ bf16 lP[4][16 * 72];   // wave-private 16x64 P tile, ld=72 (padded)
  const int tid  = threadIdx.x;
  const int w    = tid >> 5;
  const int lane = tid & 31;
  const int n16  = lane & 15;
  const int half = lane >> 4;
  const int qbase = blockIdx.x * 64 + 16 * w;
  const int bh    = blockIdx.y;                 // b*NH + h
  const bf16* __restrict__ Qb = Q  + (size_t)bh * NT * DH;
  const bf16* __restrict__ Kb = K  + (size_t)bh * NT * DH;
  const bf16* __restrict__ Vb = Vt + (size_t)bh * DH * NT;

  v16bf qf0 = frag_a_ld(Qb + (size_t)qbase * DH + 0,  DH);
  v16bf qf1 = frag_a_ld(Qb + (size_t)qbase * DH + 32, DH);

  const v8f vzero = {0.f,0.f,0.f,0.f,0.f,0.f,0.f,0.f};
  v8f o[4];
#pragma unroll
  for (int f = 0; f < 4; ++f) o[f] = vzero;
  float mrow[8], lsum[8];
#pragma unroll
  for (int r = 0; r < 8; ++r) { mrow[r] = -3.0e38f; lsum[r] = 0.f; }
  bf16* myP = &lP[w][0];

  for (int jb = 0; jb <= qbase + 15; jb += 64) {
    // speculative prefetch of the next key block (global_prefetch_b8)
    if (jb + 64 <= qbase + 15) {
      __builtin_prefetch(Kb + (size_t)(jb + 64) * DH + lane * DH, 0, 1);
      __builtin_prefetch(Vb + (size_t)(lane * 2) * NT + jb + 64, 0, 1);
    }
    // ---- S = Q K^T: hoist all 8 K fragments, then 8 WMMAs ----
    v16bf kf[8];
#pragma unroll
    for (int f = 0; f < 4; ++f) {
      kf[2 * f]     = frag_b_ld(Kb + (size_t)(jb + 16 * f) * DH + 0,  DH);
      kf[2 * f + 1] = frag_b_ld(Kb + (size_t)(jb + 16 * f) * DH + 32, DH);
    }
    v8f s[4];
#pragma unroll
    for (int f = 0; f < 4; ++f) {
      s[f] = vzero;
      s[f] = wmma_bf16(qf0, kf[2 * f],     s[f]);
      s[f] = wmma_bf16(qf1, kf[2 * f + 1], s[f]);
    }
    // ---- issue V fragment loads now; latency hides under softmax ----
    v16bf vf[8];
#pragma unroll
    for (int c2 = 0; c2 < 2; ++c2)
#pragma unroll
      for (int f = 0; f < 4; ++f)
        vf[4 * c2 + f] =
            frag_b_ld(Vb + (size_t)(16 * f) * NT + jb + 32 * c2, NT);
    // ---- causal mask (only on diagonal-straddling blocks) ----
    if (jb + 63 > qbase) {
#pragma unroll
      for (int f = 0; f < 4; ++f)
#pragma unroll
        for (int r = 0; r < 8; ++r) {
          const int qi = qbase + r + 8 * half;
          const int ji = jb + 16 * f + n16;
          if (ji > qi) s[f][r] = -3.0e38f;
        }
    }
    // ---- online softmax: row max across frags then across the 16-lane half --
    float pm[8];
#pragma unroll
    for (int r = 0; r < 8; ++r)
      pm[r] = fmaxf(fmaxf(s[0][r], s[1][r]), fmaxf(s[2][r], s[3][r]));
#pragma unroll
    for (int xm = 1; xm <= 8; xm <<= 1)
#pragma unroll
      for (int r = 0; r < 8; ++r)
        pm[r] = fmaxf(pm[r], __shfl_xor(pm[r], xm, 32));
    float scl[8];
#pragma unroll
    for (int r = 0; r < 8; ++r) {
      float mn = fmaxf(mrow[r], pm[r]);
      scl[r] = __expf(mrow[r] - mn);
      mrow[r] = mn;
    }
    float ps[8];
#pragma unroll
    for (int r = 0; r < 8; ++r) ps[r] = 0.f;
#pragma unroll
    for (int f = 0; f < 4; ++f)
#pragma unroll
      for (int r = 0; r < 8; ++r) {
        float p = __expf(s[f][r] - mrow[r]);
        s[f][r] = p;
        ps[r] += p;
      }
#pragma unroll
    for (int xm = 1; xm <= 8; xm <<= 1)
#pragma unroll
      for (int r = 0; r < 8; ++r) ps[r] += __shfl_xor(ps[r], xm, 32);
#pragma unroll
    for (int r = 0; r < 8; ++r) lsum[r] = lsum[r] * scl[r] + ps[r];
#pragma unroll
    for (int f = 0; f < 4; ++f)
#pragma unroll
      for (int r = 0; r < 8; ++r) o[f][r] *= scl[r];
    // ---- P: C-layout regs -> LDS -> A-layout frags (same-wave, in-order DS) -
#pragma unroll
    for (int f = 0; f < 4; ++f)
#pragma unroll
      for (int r = 0; r < 8; ++r)
        myP[(r + 8 * half) * 72 + 16 * f + n16] = (bf16)s[f][r];
    // ---- O += P V  (K-dim = key index, 2 chunks of 32) ----
#pragma unroll
    for (int c2 = 0; c2 < 2; ++c2) {
      v16bf ap = frag_a_ld(myP + 32 * c2, 72);
#pragma unroll
      for (int f = 0; f < 4; ++f)
        o[f] = wmma_bf16(ap, vf[4 * c2 + f], o[f]);
    }
  }
  // ---- normalize and scatter to attn [B, T, C] (f32) ----
  const int bidx = bh >> 4;
  const int hidx = bh & 15;
#pragma unroll
  for (int r = 0; r < 8; ++r) {
    const float inv = 1.0f / lsum[r];
    const int   tt  = qbase + r + 8 * half;
    const size_t rowoff = ((size_t)bidx * NT + tt) * CCH + hidx * DH;
#pragma unroll
    for (int f = 0; f < 4; ++f)
      attn[rowoff + 16 * f + n16] = o[f][r] * inv;
  }
}

// ---------------------------------------------------------------------------
// Kernel 3: y = LN(attn + x) * gamma + beta.  One block per token row.
// ---------------------------------------------------------------------------
__global__ __launch_bounds__(256) void ln_kernel(
    const float* __restrict__ x, const float* __restrict__ attn,
    const float* __restrict__ gamma, const float* __restrict__ beta,
    float* __restrict__ out) {
  __shared__ float red[256];
  const int row = blockIdx.x;
  const int tid = threadIdx.x;
  const size_t base = (size_t)row * CCH + tid * 4;
  float4 xa = *(const float4*)(x + base);
  float4 aa = *(const float4*)(attn + base);
  float v[4] = {xa.x + aa.x, xa.y + aa.y, xa.z + aa.z, xa.w + aa.w};
  red[tid] = v[0] + v[1] + v[2] + v[3];
  __syncthreads();
  for (int st = 128; st > 0; st >>= 1) {
    if (tid < st) red[tid] += red[tid + st];
    __syncthreads();
  }
  const float mu = red[0] * (1.0f / CCH);
  __syncthreads();
  float d[4]; float sq = 0.f;
#pragma unroll
  for (int e = 0; e < 4; ++e) { d[e] = v[e] - mu; sq += d[e] * d[e]; }
  red[tid] = sq;
  __syncthreads();
  for (int st = 128; st > 0; st >>= 1) {
    if (tid < st) red[tid] += red[tid + st];
    __syncthreads();
  }
  const float rs = rsqrtf(red[0] * (1.0f / CCH) + 1e-5f);
  float4 ov;
  ov.x = d[0] * rs * gamma[tid * 4 + 0] + beta[tid * 4 + 0];
  ov.y = d[1] * rs * gamma[tid * 4 + 1] + beta[tid * 4 + 1];
  ov.z = d[2] * rs * gamma[tid * 4 + 2] + beta[tid * 4 + 2];
  ov.w = d[3] * rs * gamma[tid * 4 + 3] + beta[tid * 4 + 3];
  *(float4*)(out + base) = ov;
}

extern "C" void kernel_launch(void* const* d_in, const int* in_sizes, int n_in,
                              void* d_out, int out_size, void* d_ws, size_t ws_size,
                              hipStream_t stream) {
  (void)in_sizes; (void)n_in; (void)out_size;
  const float* x     = (const float*)d_in[0];
  // d_in[1] is the boolean causal mask; causality is computed analytically.
  const float* Wq    = (const float*)d_in[2];
  const float* Wk    = (const float*)d_in[3];
  const float* Wv    = (const float*)d_in[4];
  const float* gamma = (const float*)d_in[5];
  const float* beta  = (const float*)d_in[6];

  const size_t xb_bytes = (size_t)NB * NT * CCH * sizeof(bf16);    // 16 MiB
  const size_t w_bytes  = (size_t)CCH * CCH * sizeof(bf16);        //  2 MiB
  const size_t qk_bytes = xb_bytes;                                // 16 MiB each
  const size_t at_bytes = (size_t)NB * NT * CCH * sizeof(float);   // 32 MiB
  const size_t need = xb_bytes + 3 * w_bytes + 3 * qk_bytes + at_bytes;
  if (ws_size < need) return;

  char* ws = (char*)d_ws;
  bf16*  xb = (bf16*)(ws);
  bf16*  wb = (bf16*)(ws + xb_bytes);                 // [3][1024][1024]
  bf16*  q  = (bf16*)(ws + xb_bytes + 3 * w_bytes);
  bf16*  k  = (bf16*)(ws + xb_bytes + 3 * w_bytes + qk_bytes);
  bf16*  vt = (bf16*)(ws + xb_bytes + 3 * w_bytes + 2 * qk_bytes);
  float* at = (float*)(ws + xb_bytes + 3 * w_bytes + 3 * qk_bytes);

  const int xn4 = NB * NT * CCH / 4;
  const int wn4 = CCH * CCH / 4;
  cvt_bf16_kernel<<<dim3(xn4 / 256), 256, 0, stream>>>(x, xb, xn4);
  cvt_bf16_kernel<<<dim3(wn4 / 256), 256, 0, stream>>>(Wq, wb + 0 * (size_t)CCH * CCH, wn4);
  cvt_bf16_kernel<<<dim3(wn4 / 256), 256, 0, stream>>>(Wk, wb + 1 * (size_t)CCH * CCH, wn4);
  cvt_bf16_kernel<<<dim3(wn4 / 256), 256, 0, stream>>>(Wv, wb + 2 * (size_t)CCH * CCH, wn4);

  qkv_gemm_kernel<<<dim3(CCH / 128, (NB * NT) / 64, 3), 256, 0, stream>>>(
      xb, wb, q, k, vt);
  flash_attn_kernel<<<dim3(NT / 64, NB * NH), 128, 0, stream>>>(q, k, vt, at);
  ln_kernel<<<dim3(NB * NT), 256, 0, stream>>>(x, at, gamma, beta, (float*)d_out);
}
